// DynamicConv_25761213841930
// MI455X (gfx1250) — matrix operand order repaired and checked by reading, not compiled, not run on hardware
//
#include <hip/hip_runtime.h>

typedef __attribute__((ext_vector_type(16))) __bf16 v16bf;
typedef __attribute__((ext_vector_type(8)))  float  v8f;

#define H_IMG 256
#define CI 64
#define CO 64
#define NS 16
#define KWN 4
#define TAUV 30.0f
#define TW 32
#define THR 8
#define PADW 34
#define PADH 10

__device__ __forceinline__ unsigned short f2bf(float f) {
    union { float f; unsigned int u; } v; v.f = f;
    unsigned int u = v.u;
    return (unsigned short)((u + 0x7FFFu + ((u >> 16) & 1u)) >> 16);
}

// ---- K0: zero the pooled-sum scratch ------------------------------------
__global__ void k0_zero(float* __restrict__ p, int nElems) {
    int i = blockIdx.x * 256 + threadIdx.x;
    if (i < nElems) p[i] = 0.f;
}

// ---- K1: msum[n][16] = sum_pixels relu(att_w1 @ x + b1) -----------------
__global__ void __launch_bounds__(256)
k1_att_pool(const float* __restrict__ x,
            const float* __restrict__ att_w1,
            const float* __restrict__ att_b1,
            float* __restrict__ msum) {
    __shared__ float w1[16 * CI];
    __shared__ float sacc[16];
    const int tid = threadIdx.x;
    const int n = blockIdx.y;
    for (int i = tid; i < 16 * CI; i += 256) w1[i] = att_w1[i];
    if (tid < 16) sacc[tid] = 0.f;
    __syncthreads();

    float acc[16];
#pragma unroll
    for (int d = 0; d < 16; ++d) acc[d] = 0.f;

    const int P = H_IMG * H_IMG;
    const int p0 = blockIdx.x * (P / 64);   // 1024 pixels per block
    const float* xb = x + (size_t)n * CI * P;

    for (int pp = 0; pp < 1024; pp += 256) {
        const int p = p0 + pp + tid;
        float dot[16];
#pragma unroll
        for (int d = 0; d < 16; ++d) dot[d] = 0.f;
        for (int c = 0; c < CI; ++c) {
            const float xv = xb[(size_t)c * P + p];
#pragma unroll
            for (int d = 0; d < 16; ++d) dot[d] = fmaf(w1[d * CI + c], xv, dot[d]);
        }
#pragma unroll
        for (int d = 0; d < 16; ++d) {
            const float h = dot[d] + att_b1[d];
            acc[d] += (h > 0.f) ? h : 0.f;
        }
    }
#pragma unroll
    for (int d = 0; d < 16; ++d) atomicAdd(&sacc[d], acc[d]);
    __syncthreads();
    if (tid < 16) atomicAdd(&msum[n * 16 + tid], sacc[tid]);
}

// ---- K2: pi = softmax((att_w2@mean + b2)/tau); bagg = b_bank @ pi -------
__global__ void k2_softmax_bias(const float* __restrict__ msum,
                                const float* __restrict__ att_w2,
                                const float* __restrict__ att_b2,
                                const float* __restrict__ b_bank,
                                float* __restrict__ pi,
                                float* __restrict__ bagg) {
    __shared__ float spi[NS * KWN];
    const int tid = threadIdx.x;
    if (tid < NS) {
        const int n = tid;
        float m[16];
#pragma unroll
        for (int d = 0; d < 16; ++d) m[d] = msum[n * 16 + d] * (1.0f / 65536.0f);
        float lg[KWN];
        float mx = -1e30f;
#pragma unroll
        for (int e = 0; e < KWN; ++e) {
            float s = att_b2[e];
#pragma unroll
            for (int d = 0; d < 16; ++d) s = fmaf(att_w2[e * 16 + d], m[d], s);
            s *= (1.0f / TAUV);
            lg[e] = s;
            mx = fmaxf(mx, s);
        }
        float den = 0.f;
#pragma unroll
        for (int e = 0; e < KWN; ++e) { lg[e] = __expf(lg[e] - mx); den += lg[e]; }
#pragma unroll
        for (int e = 0; e < KWN; ++e) {
            const float pv = lg[e] / den;
            spi[n * KWN + e] = pv;
            pi[n * KWN + e] = pv;
        }
    }
    __syncthreads();
    for (int i = tid; i < NS * CO; i += 256) {
        const int n = i / CO, o = i % CO;
        float s = 0.f;
#pragma unroll
        for (int k = 0; k < KWN; ++k) s = fmaf(b_bank[o * KWN + k], spi[n * KWN + k], s);
        bagg[i] = s;
    }
}

// ---- K3: aggregate weights and swizzle into WMMA A-fragment layout ------
// wfrag[n][g(4 Co-groups)][kc(18 K-chunks)][lane(32)][j(16)]  (bf16)
// K-ordering: k = (dy*3+dx)*64 + c.  A lane layout used here (16-bit A, contiguous K):
//   lanes 0-15 : M=lane,    j0..15 -> K 0..15
//   lanes 16-31: M=lane-16, j0..15 -> K 16..31
__global__ void k3_wfrag(const float* __restrict__ w_bank,
                         const float* __restrict__ pi,
                         unsigned short* __restrict__ wfrag) {
    const int idx = blockIdx.x * 256 + threadIdx.x;
    const int j = idx & 15;
    const int lane = (idx >> 4) & 31;
    const int kc = (idx >> 9) % 18;
    const int rest = (idx >> 9) / 18;
    const int g = rest & 3;
    const int n = rest >> 2;
    if (n >= NS) return;

    const int hi = lane >> 4;
    const int co = g * 16 + (lane & 15);
    const int Kc = hi * 16 + j;
    const int k = kc * 32 + Kc;
    const int s = k >> 6;     // spatial tap dy*3+dx
    const int c = k & 63;     // input channel

    float acc = 0.f;
#pragma unroll
    for (int kw = 0; kw < KWN; ++kw)
        acc = fmaf(w_bank[(((size_t)co * KWN + kw) * CI + c) * 9 + s],
                   pi[n * KWN + kw], acc);
    wfrag[idx] = f2bf(acc);
}

// ---- K4: implicit-GEMM conv via v_wmma_f32_16x16x32_bf16 ----------------
// Each wave owns 2 Co-groups x 4 pixel-groups: every B fragment (LDS) feeds
// two WMMAs, halving DS bandwidth per matrix op. B is double-buffered and A
// is register-prefetched one K-chunk ahead so ds/global waits overlap WMMA.
__global__ void __launch_bounds__(256)
k4_conv(const float* __restrict__ x,
        const unsigned short* __restrict__ wfrag,
        const float* __restrict__ bagg,
        float* __restrict__ y) {
    __shared__ alignas(32) unsigned short xt[PADH * PADW * CI]; // [y][x][c] bf16

    const int tid = threadIdx.x;
    const int n = blockIdx.z;
    const int y0 = blockIdx.y * THR;
    const int x0 = blockIdx.x * TW;
    const int P = H_IMG * H_IMG;
    const float* xb = x + (size_t)n * CI * P;

    const int wv = tid >> 5;        // wave 0..7
    const int lane = tid & 31;
    const int g0 = (wv & 1) * 2;    // this wave's Co-groups: g0, g0+1
    const int qb = (wv >> 1) * 4;   // 4 of 16 pixel-groups
    const int lo16 = lane & 15;
    const int hi = lane >> 4;

    const unsigned short* wf0 =
        wfrag + ((((size_t)n * 4 + g0) * 18) * 32 + lane) * 16;
    const unsigned short* wf1 = wf0 + (size_t)18 * 512;

    __builtin_prefetch(wf0, 0, 3);
    __builtin_prefetch(wf1, 0, 3);

    // First A fragments in flight before the staging barrier.
    v16bf a0 = *(const v16bf*)(wf0);
    v16bf a1 = *(const v16bf*)(wf1);

    // Stage halo'd input tile, fp32 -> bf16. x-fastest decode keeps global
    // loads coalesced (34-wide runs); LDS layout keeps channels contiguous.
    for (int i = tid; i < CI * PADH * PADW; i += 256) {
        const int xx = i % PADW;
        const int t = i / PADW;
        const int yy = t % PADH;
        const int c = t / PADH;
        const int gx = x0 + xx - 1;
        const int gy = y0 + yy - 1;
        float v = 0.f;
        if ((unsigned)gx < H_IMG && (unsigned)gy < H_IMG)
            v = xb[(size_t)c * P + gy * H_IMG + gx];
        xt[(yy * PADW + xx) * CI + c] = f2bf(v);
    }
    __syncthreads();

    // Precompute per-pixel-group LDS base addresses (elements).
    int baseq[4];
#pragma unroll
    for (int i = 0; i < 4; ++i) {
        const int q = qb + i;
        const int r = q >> 1;
        const int px = ((q & 1) << 4) + lo16;
        baseq[i] = (r * PADW + px) * CI + (hi << 4);
    }

    v8f acc0[4] = {};
    v8f acc1[4] = {};

    for (int kc = 0; kc < 18; ++kc) {
        // Prefetch next A fragments (wrap to 0 on last iter: in-bounds, cheap).
        const int kcn = (kc + 1 < 18) ? (kc + 1) : 0;
        const v16bf a0n = *(const v16bf*)(wf0 + (size_t)kcn * 512);
        const v16bf a1n = *(const v16bf*)(wf1 + (size_t)kcn * 512);

        const int s = kc >> 1;
        const int koff = ((s / 3) * PADW + (s % 3)) * CI + ((kc & 1) << 5);

        v16bf bc = *(const v16bf*)(&xt[baseq[0] + koff]);
#pragma unroll
        for (int i = 0; i < 4; ++i) {
            const v16bf bn =
                (i < 3) ? *(const v16bf*)(&xt[baseq[i + 1] + koff]) : bc;
            acc0[i] = __builtin_amdgcn_wmma_f32_16x16x32_bf16(
                false, a0, false, bc, (short)0, acc0[i], false, false);
            acc1[i] = __builtin_amdgcn_wmma_f32_16x16x32_bf16(
                false, a1, false, bc, (short)0, acc1[i], false, false);
            bc = bn;
        }
        a0 = a0n;
        a1 = a1n;
    }

    // Bias + store. D layout: VGPR v, lane half -> M = v + 8*hi, N = lane%16.
    float bco0[8], bco1[8];
#pragma unroll
    for (int v = 0; v < 8; ++v) {
        bco0[v] = bagg[n * CO + g0 * 16 + hi * 8 + v];
        bco1[v] = bagg[n * CO + (g0 + 1) * 16 + hi * 8 + v];
    }

    float* yb = y + (size_t)n * CO * P;
#pragma unroll
    for (int i = 0; i < 4; ++i) {
        const int q = qb + i;
        const int r = q >> 1;
        const int gx = x0 + ((q & 1) << 4) + lo16;
        const int gy = y0 + r;
        float* yp = yb + (size_t)gy * H_IMG + gx;
#pragma unroll
        for (int v = 0; v < 8; ++v) {
            const int co0 = g0 * 16 + hi * 8 + v;
            yp[(size_t)co0 * P] = acc0[i][v] + bco0[v];
            yp[(size_t)(co0 + 16) * P] = acc1[i][v] + bco1[v];
        }
    }
}

extern "C" void kernel_launch(void* const* d_in, const int* in_sizes, int n_in,
                              void* d_out, int out_size, void* d_ws, size_t ws_size,
                              hipStream_t stream) {
    const float* x      = (const float*)d_in[0];
    const float* w_bank = (const float*)d_in[1];
    const float* b_bank = (const float*)d_in[2];
    const float* att_w1 = (const float*)d_in[3];
    const float* att_b1 = (const float*)d_in[4];
    const float* att_w2 = (const float*)d_in[5];
    const float* att_b2 = (const float*)d_in[6];
    float* out = (float*)d_out;

    char* ws = (char*)d_ws;
    float* msum = (float*)ws;                              // 256 f32
    float* pi   = (float*)(ws + 1024);                     // 64 f32
    float* bagg = (float*)(ws + 2048);                     // 1024 f32
    unsigned short* wfrag = (unsigned short*)(ws + 8192);  // 589824 bf16

    k0_zero<<<1, 256, 0, stream>>>(msum, 256);
    k1_att_pool<<<dim3(64, 16), 256, 0, stream>>>(x, att_w1, att_b1, msum);
    k2_softmax_bias<<<1, 256, 0, stream>>>(msum, att_w2, att_b2, b_bank, pi, bagg);
    k3_wfrag<<<2304, 256, 0, stream>>>(w_bank, pi, wfrag);
    k4_conv<<<dim3(H_IMG / TW, H_IMG / THR, NS), 256, 0, stream>>>(x, wfrag, bagg, out);
}
